// PointNet2Encoder_87582973100180
// MI455X (gfx1250) — compile-verified, matrix operand order checked by reading, not run on hardware
//
#include <hip/hip_runtime.h>
#include <hip/hip_bf16.h>

typedef __attribute__((ext_vector_type(16))) _Float16 v16h;
typedef __attribute__((ext_vector_type(8)))  _Float16 v8h;
typedef __attribute__((ext_vector_type(8)))  float    v8f;

#define EPSN 1e-5f

// ------------------------------------------------------------------
// CDNA5 async global->LDS copy (ASYNCcnt-tracked), per-lane b128.
// ------------------------------------------------------------------
__device__ __forceinline__ void async_ld_b128(unsigned lds_off, const void* gaddr)
{
    unsigned long long ga = (unsigned long long)(uintptr_t)gaddr;
    asm volatile("global_load_async_to_lds_b128 %0, %1, off"
                 :: "v"(lds_off), "v"(ga) : "memory");
}
__device__ __forceinline__ void wait_async0()
{
    asm volatile("s_wait_asynccnt 0" ::: "memory");
}

// Build a v16h fragment from one 64B LDS row: two contiguous 16B runs per lane
// (A/B 16-bit WMMA layout: e<8 -> K=8*hi+e ; e>=8 -> K=16+8*hi+(e-8)).
__device__ __forceinline__ v16h frag16(const _Float16* rowp, int hi)
{
    v8h lo = *(const v8h*)(rowp + 8 * hi);
    v8h hh = *(const v8h*)(rowp + 16 + 8 * hi);
    return __builtin_shufflevector(lo, hh, 0,1,2,3,4,5,6,7,8,9,10,11,12,13,14,15);
}

// ------------------------------------------------------------------
// Farthest point sampling: one block per batch element.
// ------------------------------------------------------------------
__global__ void fps_kernel(const float* __restrict__ xyz, int* __restrict__ fidx,
                           float* __restrict__ dists, int Npts, int S)
{
    __shared__ float sval[256];
    __shared__ int   sidx[256];
    __shared__ int   s_last;
    const int b   = blockIdx.x;
    const int tid = threadIdx.x;
    const float* pts = xyz + (size_t)b * Npts * 3;
    float* dd = dists + (size_t)b * Npts;
    for (int i = tid; i < Npts; i += blockDim.x) dd[i] = 1e10f;
    if (tid == 0) { fidx[(size_t)b * S] = 0; s_last = 0; }
    __syncthreads();
    for (int s = 1; s < S; ++s) {
        const int last = s_last;
        const float lx = pts[last*3+0], ly = pts[last*3+1], lz = pts[last*3+2];
        float best = -1.0f; int bi = 0;
        for (int i = tid; i < Npts; i += blockDim.x) {
            float dx = pts[i*3+0]-lx, dy = pts[i*3+1]-ly, dz = pts[i*3+2]-lz;
            float d  = dx*dx + dy*dy + dz*dz;
            float m  = fminf(dd[i], d);
            dd[i] = m;
            if (m > best) { best = m; bi = i; }
        }
        sval[tid] = best; sidx[tid] = bi;
        __syncthreads();
        for (int off = 128; off > 0; off >>= 1) {
            if (tid < off && sval[tid+off] > sval[tid]) {
                sval[tid] = sval[tid+off]; sidx[tid] = sidx[tid+off];
            }
            __syncthreads();
        }
        if (tid == 0) { s_last = sidx[0]; fidx[(size_t)b*S + s] = sidx[0]; }
        __syncthreads();
    }
}

__global__ void gather_xyz_kernel(const float* __restrict__ xyz, const int* __restrict__ fidx,
                                  float* __restrict__ nxyz, int Npts, int S, int B)
{
    int q = blockIdx.x * blockDim.x + threadIdx.x;
    if (q >= B * S) return;
    int b = q / S;
    int p = fidx[q];
    const float* src = xyz + ((size_t)b * Npts + p) * 3;
    nxyz[(size_t)q*3+0] = src[0];
    nxyz[(size_t)q*3+1] = src[1];
    nxyz[(size_t)q*3+2] = src[2];
}

// ------------------------------------------------------------------
// Ball query: first nsample points (index order) within radius; pad with
// the first found index.
// ------------------------------------------------------------------
__global__ void ball_query_kernel(const float* __restrict__ xyz, const float* __restrict__ nxyz,
                                  int* __restrict__ idx, int B, int Npts, int S, int ns, float r2)
{
    int q = blockIdx.x * blockDim.x + threadIdx.x;
    if (q >= B * S) return;
    int b = q / S;
    const float qx = nxyz[(size_t)q*3+0], qy = nxyz[(size_t)q*3+1], qz = nxyz[(size_t)q*3+2];
    const float* pts = xyz + (size_t)b * Npts * 3;
    int* o = idx + (size_t)q * ns;
    int cnt = 0, first = 0; bool have = false;
    for (int i = 0; i < Npts; ++i) {
        float dx = pts[i*3+0]-qx, dy = pts[i*3+1]-qy, dz = pts[i*3+2]-qz;
        if (dx*dx + dy*dy + dz*dz < r2) {
            if (!have) { first = i; have = true; }
            o[cnt++] = i;
            if (cnt >= ns) break;
        }
    }
    for (int k = cnt; k < ns; ++k) o[k] = first;
}

// ------------------------------------------------------------------
// Grouping: build f16 activation rows [centered_xyz | feats | 0-pad].
// ------------------------------------------------------------------
__global__ void group_kernel(const float* __restrict__ xyz, const _Float16* __restrict__ feats,
                             const int* __restrict__ idx, const float* __restrict__ nxyz,
                             _Float16* __restrict__ Xout,
                             int B, int Npts, int S, int ns, int C, int Kpad)
{
    long long m = (long long)blockIdx.x * blockDim.x + threadIdx.x;
    long long M = (long long)B * S * ns;
    if (m >= M) return;
    int b = (int)(m / ((long long)S * ns));
    int rem = (int)(m % ((long long)S * ns));
    int s = rem / ns, j = rem % ns;
    int p = idx ? idx[m] : j;
    const float* pt = xyz + ((size_t)b * Npts + p) * 3;
    float cx = 0.f, cy = 0.f, cz = 0.f;
    if (nxyz) {
        const float* c = nxyz + ((size_t)b * S + s) * 3;
        cx = c[0]; cy = c[1]; cz = c[2];
    }
    _Float16* o = Xout + (size_t)m * Kpad;
    o[0] = (_Float16)(pt[0] - cx);
    o[1] = (_Float16)(pt[1] - cy);
    o[2] = (_Float16)(pt[2] - cz);
    if (feats) {
        const _Float16* f = feats + ((size_t)b * Npts + p) * C;
        for (int c = 0; c < C; ++c) o[3 + c] = f[c];
    }
    for (int k = 3 + C; k < Kpad; ++k) o[k] = (_Float16)0.f;
}

// Weight prep: W (Cout x Cin) f32 -> Wt (Cout x Kpad) f16, zero-padded cols.
// (Row-major Cout x K is the B^T layout the LDS-staged GEMM consumes.)
__global__ void wprep_kernel(const float* __restrict__ W, _Float16* __restrict__ Wt,
                             int Cin, int Cout, int Kpad)
{
    int i = blockIdx.x * blockDim.x + threadIdx.x;
    if (i >= Kpad * Cout) return;
    int n = i / Kpad, k = i % Kpad;
    Wt[i] = (k < Cin) ? (_Float16)W[(size_t)n * Cin + k] : (_Float16)0.f;
}

__global__ void zero_kernel(float* __restrict__ p, int n)
{
    int i = blockIdx.x * blockDim.x + threadIdx.x;
    if (i < n) p[i] = 0.f;
}

// ------------------------------------------------------------------
// LDS-staged WMMA GEMM: Y(MxNc,f32) = X(MxK,f16) * Wt(NcxK,f16)^T,
// plus per-channel sum/sumsq into stats[2*Nc].
// Block tile 128x64, 8 waves, each wave = 2x2 grid of 16x16 WMMA tiles.
// A/B tiles staged via global_load_async_to_lds_b128 (ASYNCcnt).
// Grid is exact (M%128==0, Nc%64==0, K%32==0) so EXEC stays all-ones.
// ------------------------------------------------------------------
__global__ void __launch_bounds__(256)
wmma_gemm_stats(const _Float16* __restrict__ X,
                const _Float16* __restrict__ Wt,
                float* __restrict__ Y,
                float* __restrict__ stats,
                int M, int K, int Nc)
{
    __shared__ __align__(16) _Float16 As[128][32];
    __shared__ __align__(16) _Float16 Bs[64][32];

    const int t    = threadIdx.x;
    const int wave = t >> 5;
    const int lane = t & 31;
    const int nb   = Nc >> 6;
    const int bm   = blockIdx.x / nb;
    const int bn   = blockIdx.x % nb;
    const int m0   = bm * 128;
    const int n0   = bn * 64;

    const int wm  = wave & 3;      // M subgroup (32 rows)
    const int wn  = wave >> 2;     // N subgroup (32 cols)
    const int row = lane & 15;
    const int hi  = lane >> 4;

    // staging chunk coords: 16B (8-half) chunks
    const int ar0  = t >> 2;               // A rows 0..63
    const int ar1  = ar0 + 64;             // A rows 64..127
    const int aoff = (t & 3) * 8;
    const int bnr  = t >> 2;               // B rows 0..63
    const int boff = (t & 3) * 8;

    v8f acc00 = {}, acc01 = {}, acc10 = {}, acc11 = {};

    for (int k0 = 0; k0 < K; k0 += 32) {
        __syncthreads();                                    // previous compute done
        async_ld_b128((unsigned)(uintptr_t)&As[ar0][aoff],
                      X + (size_t)(m0 + ar0) * K + k0 + aoff);
        async_ld_b128((unsigned)(uintptr_t)&As[ar1][aoff],
                      X + (size_t)(m0 + ar1) * K + k0 + aoff);
        async_ld_b128((unsigned)(uintptr_t)&Bs[bnr][boff],
                      Wt + (size_t)(n0 + bnr) * K + k0 + boff);
        wait_async0();
        __syncthreads();                                    // tiles visible

        v16h a0 = frag16(&As[wm*32 +      row][0], hi);
        v16h a1 = frag16(&As[wm*32 + 16 + row][0], hi);
        v16h b0 = frag16(&Bs[wn*32 +      row][0], hi);     // row == B column idx
        v16h b1 = frag16(&Bs[wn*32 + 16 + row][0], hi);

        acc00 = __builtin_amdgcn_wmma_f32_16x16x32_f16(false, a0, false, b0, (short)0, acc00, false, false);
        acc01 = __builtin_amdgcn_wmma_f32_16x16x32_f16(false, a0, false, b1, (short)0, acc01, false, false);
        acc10 = __builtin_amdgcn_wmma_f32_16x16x32_f16(false, a1, false, b0, (short)0, acc10, false, false);
        acc11 = __builtin_amdgcn_wmma_f32_16x16x32_f16(false, a1, false, b1, (short)0, acc11, false, false);
    }

    const int cm0 = m0 + wm * 32;
    const int cn0 = n0 + wn * 32;
    float s0 = 0.f, ss0 = 0.f, s1 = 0.f, ss1 = 0.f;
#pragma unroll
    for (int v = 0; v < 8; ++v) {
        const int mA = cm0 + v + 8 * hi;
        const int mB = mA + 16;
        const int nA = cn0 + row;
        const int nB = nA + 16;
        float y;
        y = acc00[v]; Y[(size_t)mA * Nc + nA] = y; s0 += y; ss0 += y * y;
        y = acc10[v]; Y[(size_t)mB * Nc + nA] = y; s0 += y; ss0 += y * y;
        y = acc01[v]; Y[(size_t)mA * Nc + nB] = y; s1 += y; ss1 += y * y;
        y = acc11[v]; Y[(size_t)mB * Nc + nB] = y; s1 += y; ss1 += y * y;
    }
    // lanes L and L+16 share an output column: fold, then lanes 0..15 atomically
    // accumulate full 32-row column partials.
    s0 += __shfl_xor(s0, 16, 32); ss0 += __shfl_xor(ss0, 16, 32);
    s1 += __shfl_xor(s1, 16, 32); ss1 += __shfl_xor(ss1, 16, 32);
    if (hi == 0) {
        atomicAdd(&stats[cn0 + lane],           s0);
        atomicAdd(&stats[Nc + cn0 + lane],      ss0);
        atomicAdd(&stats[cn0 + 16 + lane],      s1);
        atomicAdd(&stats[Nc + cn0 + 16 + lane], ss1);
    }
}

// BatchNorm (batch stats) + ReLU, repack f16 with zero-padded stride.
__global__ void bn_relu_kernel(const float* __restrict__ Y, const float* __restrict__ stats,
                               const float* __restrict__ gamma, const float* __restrict__ beta,
                               _Float16* __restrict__ Xout, int M, int Nc, int strideOut)
{
    long long i = (long long)blockIdx.x * blockDim.x + threadIdx.x;
    long long total = (long long)M * strideOut;
    if (i >= total) return;
    int n = (int)(i % strideOut);
    long long m = i / strideOut;
    if (n < Nc) {
        float invM = 1.0f / (float)M;
        float mean = stats[n] * invM;
        float var  = stats[Nc + n] * invM - mean * mean;
        float y = (Y[(size_t)m * Nc + n] - mean) * rsqrtf(var + EPSN) * gamma[n] + beta[n];
        Xout[i] = (_Float16)fmaxf(y, 0.f);
    } else {
        Xout[i] = (_Float16)0.f;
    }
}

// Max over the nsample axis: out[p,c] = max_j X[(p*ns+j)*Nc + c]
__global__ void maxpool_kernel(const _Float16* __restrict__ X, _Float16* __restrict__ out,
                               int P, int ns, int Nc)
{
    long long i = (long long)blockIdx.x * blockDim.x + threadIdx.x;
    if (i >= (long long)P * Nc) return;
    int p = (int)(i / Nc), c = (int)(i % Nc);
    const _Float16* base = X + ((size_t)p * ns) * Nc + c;
    float m = -3.4e38f;
    for (int j = 0; j < ns; ++j) m = fmaxf(m, (float)base[(size_t)j * Nc]);
    out[i] = (_Float16)m;
}

__global__ void h2f_kernel(const _Float16* __restrict__ x, float* __restrict__ y, int n)
{
    int i = blockIdx.x * blockDim.x + threadIdx.x;
    if (i < n) y[i] = (float)x[i];
}

// FC (+ optional LayerNorm + ReLU). One block per batch row (B=32).
__global__ void fc_ln_relu_kernel(const float* __restrict__ x, const float* __restrict__ W,
                                  const float* __restrict__ bias,
                                  const float* __restrict__ lng, const float* __restrict__ lnb,
                                  float* __restrict__ out, int K, int Nc, int doLnRelu)
{
    __shared__ float sy[512];
    __shared__ float red[256];
    const int b = blockIdx.x, tid = threadIdx.x;
    const float* xr = x + (size_t)b * K;
    for (int n = tid; n < Nc; n += blockDim.x) {
        const float* wr = W + (size_t)n * K;
        float acc = bias[n];
        for (int k = 0; k < K; ++k) acc += xr[k] * wr[k];
        sy[n] = acc;
    }
    __syncthreads();
    if (!doLnRelu) {
        for (int n = tid; n < Nc; n += blockDim.x) out[(size_t)b * Nc + n] = sy[n];
        return;
    }
    float loc = 0.f;
    for (int n = tid; n < Nc; n += blockDim.x) loc += sy[n];
    red[tid] = loc; __syncthreads();
    for (int off = 128; off > 0; off >>= 1) {
        if (tid < off) red[tid] += red[tid + off];
        __syncthreads();
    }
    float mean = red[0] / (float)Nc;
    __syncthreads();
    loc = 0.f;
    for (int n = tid; n < Nc; n += blockDim.x) { float d = sy[n] - mean; loc += d * d; }
    red[tid] = loc; __syncthreads();
    for (int off = 128; off > 0; off >>= 1) {
        if (tid < off) red[tid] += red[tid + off];
        __syncthreads();
    }
    float inv = rsqrtf(red[0] / (float)Nc + EPSN);
    for (int n = tid; n < Nc; n += blockDim.x)
        out[(size_t)b * Nc + n] = fmaxf((sy[n] - mean) * inv * lng[n] + lnb[n], 0.f);
}

// ------------------------------------------------------------------
// Host side
// ------------------------------------------------------------------
static inline int cdiv(long long a, int b) { return (int)((a + b - 1) / b); }

static void run_layer(const _Float16* Xin, int M, int Kpad,
                      const float* W, const float* gamma, const float* beta,
                      int Cin, int Cout,
                      _Float16* Wt, float* Ybuf, float* stats,
                      _Float16* Xout, int strideOut, hipStream_t stream)
{
    wprep_kernel<<<cdiv((long long)Kpad * Cout, 256), 256, 0, stream>>>(W, Wt, Cin, Cout, Kpad);
    zero_kernel<<<cdiv(2 * Cout, 256), 256, 0, stream>>>(stats, 2 * Cout);
    int blocks = (M / 128) * (Cout / 64);
    wmma_gemm_stats<<<blocks, 256, 0, stream>>>(Xin, Wt, Ybuf, stats, M, Kpad, Cout);
    bn_relu_kernel<<<cdiv((long long)M * strideOut, 256), 256, 0, stream>>>(
        Ybuf, stats, gamma, beta, Xout, M, Cout, strideOut);
}

extern "C" void kernel_launch(void* const* d_in, const int* in_sizes, int n_in,
                              void* d_out, int out_size, void* d_ws, size_t ws_size,
                              hipStream_t stream)
{
    (void)in_sizes; (void)n_in; (void)out_size; (void)ws_size;
    constexpr int B = 32, N = 4096;
    constexpr int S1 = 512, NS1 = 32;
    constexpr int S2 = 128, NS2 = 64;
    constexpr int M1 = B * S1 * NS1;   // 524288
    constexpr int M2 = B * S2 * NS2;   // 262144
    constexpr int M3 = B * 128;        // 4096

    const float* xyz = (const float*)d_in[0];
    const float* W1a = (const float*)d_in[1];  const float* g1a = (const float*)d_in[2];  const float* b1a = (const float*)d_in[3];
    const float* W1b = (const float*)d_in[4];  const float* g1b = (const float*)d_in[5];  const float* b1b = (const float*)d_in[6];
    const float* W1c = (const float*)d_in[7];  const float* g1c = (const float*)d_in[8];  const float* b1c = (const float*)d_in[9];
    const float* W2a = (const float*)d_in[10]; const float* g2a = (const float*)d_in[11]; const float* b2a = (const float*)d_in[12];
    const float* W2b = (const float*)d_in[13]; const float* g2b = (const float*)d_in[14]; const float* b2b = (const float*)d_in[15];
    const float* W2c = (const float*)d_in[16]; const float* g2c = (const float*)d_in[17]; const float* b2c = (const float*)d_in[18];
    const float* W3a = (const float*)d_in[19]; const float* g3a = (const float*)d_in[20]; const float* b3a = (const float*)d_in[21];
    const float* W3b = (const float*)d_in[22]; const float* g3b = (const float*)d_in[23]; const float* b3b = (const float*)d_in[24];
    const float* W3c = (const float*)d_in[25]; const float* g3c = (const float*)d_in[26]; const float* b3c = (const float*)d_in[27];
    const float* fc1w = (const float*)d_in[28]; const float* fc1b = (const float*)d_in[29];
    const float* ln1g = (const float*)d_in[30]; const float* ln1b = (const float*)d_in[31];
    const float* fc2w = (const float*)d_in[32]; const float* fc2b = (const float*)d_in[33];
    const float* ln2g = (const float*)d_in[34]; const float* ln2b = (const float*)d_in[35];
    const float* fc3w = (const float*)d_in[36]; const float* fc3b = (const float*)d_in[37];

    char* cur = (char*)d_ws;
    auto take = [&](size_t bytes) -> char* {
        char* r = cur;
        cur += (bytes + 255) & ~((size_t)255);
        return r;
    };
    float*    dists  = (float*)take((size_t)B * N * 4);
    int*      fidx1  = (int*)  take((size_t)B * S1 * 4);
    float*    nxyz1  = (float*)take((size_t)B * S1 * 3 * 4);
    int*      idx1   = (int*)  take((size_t)B * S1 * NS1 * 4);
    int*      fidx2  = (int*)  take((size_t)B * S2 * 4);
    float*    nxyz2  = (float*)take((size_t)B * S2 * 3 * 4);
    int*      idx2   = (int*)  take((size_t)B * S2 * NS2 * 4);
    _Float16* feats1 = (_Float16*)take((size_t)B * S1 * 128 * 2);
    _Float16* feats2 = (_Float16*)take((size_t)B * S2 * 256 * 2);
    _Float16* gh     = (_Float16*)take((size_t)B * 1024 * 2);
    float*    gf     = (float*)take((size_t)B * 1024 * 4);
    float*    h1     = (float*)take((size_t)B * 512 * 4);
    float*    h2     = (float*)take((size_t)B * 256 * 4);
    float*    stats  = (float*)take((size_t)2 * 1024 * 4);
    _Float16* Wt     = (_Float16*)take((size_t)1024 * 512 * 2);
    _Float16* Xa     = (_Float16*)take((size_t)M2 * 160 * 2);   // >= M1*64
    _Float16* Xb     = (_Float16*)take((size_t)M2 * 256 * 2);   // == M1*128
    float*    Ybuf   = (float*)take((size_t)M1 * 128 * 4);      // == M2*256

    // ================= SA1 =================
    fps_kernel<<<B, 256, 0, stream>>>(xyz, fidx1, dists, N, S1);
    gather_xyz_kernel<<<cdiv(B * S1, 256), 256, 0, stream>>>(xyz, fidx1, nxyz1, N, S1, B);
    ball_query_kernel<<<cdiv(B * S1, 256), 256, 0, stream>>>(xyz, nxyz1, idx1, B, N, S1, NS1, 0.2f * 0.2f);
    group_kernel<<<cdiv((long long)M1, 256), 256, 0, stream>>>(xyz, nullptr, idx1, nxyz1, Xa, B, N, S1, NS1, 0, 32);
    run_layer(Xa, M1, 32, W1a, g1a, b1a, 3, 64,   Wt, Ybuf, stats, Xb, 64,  stream);
    run_layer(Xb, M1, 64, W1b, g1b, b1b, 64, 64,  Wt, Ybuf, stats, Xa, 64,  stream);
    run_layer(Xa, M1, 64, W1c, g1c, b1c, 64, 128, Wt, Ybuf, stats, Xb, 128, stream);
    maxpool_kernel<<<cdiv((long long)B * S1 * 128, 256), 256, 0, stream>>>(Xb, feats1, B * S1, NS1, 128);

    // ================= SA2 =================
    fps_kernel<<<B, 256, 0, stream>>>(nxyz1, fidx2, dists, S1, S2);
    gather_xyz_kernel<<<cdiv(B * S2, 256), 256, 0, stream>>>(nxyz1, fidx2, nxyz2, S1, S2, B);
    ball_query_kernel<<<cdiv(B * S2, 256), 256, 0, stream>>>(nxyz1, nxyz2, idx2, B, S1, S2, NS2, 0.4f * 0.4f);
    group_kernel<<<cdiv((long long)M2, 256), 256, 0, stream>>>(nxyz1, feats1, idx2, nxyz2, Xa, B, S1, S2, NS2, 128, 160);
    run_layer(Xa, M2, 160, W2a, g2a, b2a, 131, 128, Wt, Ybuf, stats, Xb, 128, stream);
    run_layer(Xb, M2, 128, W2b, g2b, b2b, 128, 128, Wt, Ybuf, stats, Xa, 128, stream);
    run_layer(Xa, M2, 128, W2c, g2c, b2c, 128, 256, Wt, Ybuf, stats, Xb, 256, stream);
    maxpool_kernel<<<cdiv((long long)B * S2 * 256, 256), 256, 0, stream>>>(Xb, feats2, B * S2, NS2, 256);

    // ================= SA3 (group-all) =================
    group_kernel<<<cdiv((long long)M3, 256), 256, 0, stream>>>(nxyz2, feats2, nullptr, nullptr, Xa, B, 128, 1, 128, 256, 288);
    run_layer(Xa, M3, 288, W3a, g3a, b3a, 259, 256,  Wt, Ybuf, stats, Xb, 256,  stream);
    run_layer(Xb, M3, 256, W3b, g3b, b3b, 256, 512,  Wt, Ybuf, stats, Xa, 512,  stream);
    run_layer(Xa, M3, 512, W3c, g3c, b3c, 512, 1024, Wt, Ybuf, stats, Xb, 1024, stream);
    maxpool_kernel<<<cdiv((long long)B * 1024, 256), 256, 0, stream>>>(Xb, gh, B, 128, 1024);

    // ================= FC head =================
    h2f_kernel<<<cdiv(B * 1024, 256), 256, 0, stream>>>(gh, gf, B * 1024);
    fc_ln_relu_kernel<<<B, 256, 0, stream>>>(gf, fc1w, fc1b, ln1g, ln1b, h1, 1024, 512, 1);
    fc_ln_relu_kernel<<<B, 256, 0, stream>>>(h1, fc2w, fc2b, ln2g, ln2b, h2, 512, 256, 1);
    fc_ln_relu_kernel<<<B, 256, 0, stream>>>(h2, fc3w, fc3b, nullptr, nullptr, (float*)d_out, 256, 64, 0);
}
